// Attn_14920716386547
// MI455X (gfx1250) — matrix operand order repaired and checked by reading, not compiled, not run on hardware
//
#include <hip/hip_runtime.h>
#include <hip/hip_bf16.h>

// MI455X (gfx1250) causal multi-head attention.
// - All matmuls on V_WMMA_F32_16X16X32_F16 (f16 in, f32 accumulate).
// - QKV / output projections: 4-wave blocks, 128x64 block tile; the shared
//   64x32 weight tile is staged into LDS by the Tensor Data Mover
//   (tensor_load_to_lds, double-buffered, pipelined with s_wait_tensorcnt 1).
// - Attention: flash-style streaming softmax; logits computed transposed so
//   softmax'ed P packs directly into the A-fragment of the P x V WMMA.

typedef __attribute__((ext_vector_type(16))) _Float16 v16h;
typedef __attribute__((ext_vector_type(8)))  _Float16 v8h;
typedef __attribute__((ext_vector_type(8)))  float    v8f;
typedef __attribute__((ext_vector_type(4)))  unsigned int u32x4;
typedef __attribute__((ext_vector_type(8)))  unsigned int u32x8;

constexpr int cB = 2, cS = 2048, cD = 1024, cH = 16, cDH = 64;
constexpr int cM = cB * cS; // 4096 total rows

__device__ __forceinline__ v8f wmma16(v16h a, v16h b, v8f c) {
  // (neg_a, A, neg_b, B, c_mod, C, reuse_a, reuse_b)
  return __builtin_amdgcn_wmma_f32_16x16x32_f16(false, a, false, b, (short)0, c,
                                                false, false);
}

// A-fragment (16x32 f16): lane holds row (lane&15); the 16 K-values per lane
// are 16 contiguous halves at k0+16h.
__device__ __forceinline__ v16h load_a_frag(const _Float16* __restrict__ row,
                                            int k0, int hh) {
  v8h x0 = *(const v8h*)(row + k0 + 16 * hh);
  v8h x1 = *(const v8h*)(row + k0 + 16 * hh + 8);
  v16h a;
#pragma unroll
  for (int i = 0; i < 8; ++i) { a[i] = x0[i]; a[8 + i] = x1[i]; }
  return a;
}

// ---------------------------------------------------------------------------
// Tensor Data Mover: async-load a 2D tile (rows x 32 halves, row stride in
// elements) from global memory into LDS at lds_dst.
// D# layout per CDNA5 ISA (groups 0 and 1; VADDR2/3
// omitted -> NULL for <=2D tensors). tensor_dim0/1 are set huge so the OOB
// clamp never drops tile data. Tracked with TENSORcnt.
//
// D#.lds_addr: for flat pointers into LDS the low 32 bits are the wave-
// relative LDS byte address, so we derive it
// from the pointer itself. The pointer is ALSO passed as an (unused) "v"
// operand so its provenance escapes into the asm: with the "memory" clobber
// the compiler must assume the asm writes the LDS tile, keeping the
// subsequent ds_load_b128 reads alive (otherwise they fold to undef).
// ---------------------------------------------------------------------------
__device__ __forceinline__ void tdm_load_tile(const _Float16* lds_dst,
                                              const _Float16* gsrc,
                                              unsigned rows,
                                              unsigned row_stride_elems) {
  unsigned long long ga = (unsigned long long)(size_t)gsrc;
  unsigned lds_off = (unsigned)(size_t)lds_dst; // wave-relative LDS byte addr
  u32x4 g0;
  g0[0] = 1u;                                   // count=1 (valid user D#)
  g0[1] = lds_off;                              // lds_addr (bytes)
  g0[2] = (unsigned)(ga & 0xffffffffu);         // global_addr[31:0]
  g0[3] = (unsigned)((ga >> 32) & 0x01ffffffu)  // global_addr[56:32]
        | (2u << 30);                           // type = 2 ("image")
  u32x8 g1;
  g1[0] = (1u << 16);                           // data_size=1 (2B), wg_mask=0
  g1[1] = 0u;                                   // tensor_dim0[15:0]=0
  g1[2] = 0x4000u;                              // tensor_dim0 = 0x40000000
  g1[3] = 0x4000u | (32u << 16);                // tensor_dim1 huge, tile_dim0=32
  g1[4] = rows & 0xffffu;                       // tile_dim1 = rows, tile_dim2=0
  g1[5] = row_stride_elems;                     // tensor_dim0_stride[31:0]
  g1[6] = 0u;                                   // stride hi / dim1_stride lo
  g1[7] = 0u;
  asm volatile("tensor_load_to_lds %0, %1"
               :
               : "s"(g0), "s"(g1), "v"(lds_dst)
               : "memory");
}

__global__ void f32_to_f16(const float* __restrict__ src,
                           _Float16* __restrict__ dst, int n) {
  int i = blockIdx.x * blockDim.x + threadIdx.x;
  if (i < n) dst[i] = (_Float16)src[i];
}

// ---------------------------------------------------------------------------
// QKV projection: 4 waves / block, block tile 128(M) x 64(N); wave w owns
// rows mblk+32w..+32. B (weight) tile 64x32 staged in LDS by TDM, double-
// buffered. blockIdx.z selects Q/K/V.
// ---------------------------------------------------------------------------
__global__ __launch_bounds__(128) void qkv_gemm(
    const _Float16* __restrict__ Xh,
    const _Float16* __restrict__ Wqh, const _Float16* __restrict__ Wkh,
    const _Float16* __restrict__ Wvh,
    _Float16* __restrict__ Qo, _Float16* __restrict__ Ko,
    _Float16* __restrict__ Vt) {
  __shared__ alignas(64) _Float16 bsm[2 * 64 * 32]; // two 4 KB B tiles

  const int which = blockIdx.z;
  const _Float16* __restrict__ W = (which == 0) ? Wqh : (which == 1 ? Wkh : Wvh);
  const int mblk = blockIdx.x * 128, n0 = blockIdx.y * 64;
  const int wave = threadIdx.x >> 5, l = threadIdx.x & 31;
  const int lm = l & 15, hh = l >> 4;
  const int m0 = mblk + wave * 32;
  const _Float16* __restrict__ arow0 = Xh + (size_t)(m0 + lm) * cD;
  const _Float16* __restrict__ arow1 = Xh + (size_t)(m0 + 16 + lm) * cD;
  const bool issuer = (threadIdx.x < 32);
  const _Float16* wtile = W + (size_t)n0 * cD;

  if (issuer) tdm_load_tile(bsm, wtile, 64u, (unsigned)cD); // prologue: tile 0

  v8f acc[8] = {}; // acc[2*t + u] : N-subtile t, row-half u
  constexpr int KSTEPS = cD / 32;
  for (int i = 0; i < KSTEPS; ++i) {
    const int k0 = i * 32;
    if (issuer) {
      if (i + 1 < KSTEPS) { // issue next tile, keep <=1 in flight beyond current
        tdm_load_tile(bsm + (size_t)((i + 1) & 1) * 2048, wtile + k0 + 32, 64u,
                      (unsigned)cD);
        __builtin_amdgcn_s_wait_tensorcnt(1);
      } else {
        __builtin_amdgcn_s_wait_tensorcnt(0);
      }
    }
    __syncthreads(); // current buffer ready for all waves

    const _Float16* bs = bsm + (size_t)(i & 1) * 2048;
    if (k0 + 32 < cD) __builtin_prefetch(arow0 + k0 + 32, 0, 1);
    v16h a0 = load_a_frag(arow0, k0, hh);
    v16h a1 = load_a_frag(arow1, k0, hh);
#pragma unroll
    for (int t = 0; t < 4; ++t) {
      // B-fragment (32x16): lane = out column, 16 contiguous K halves at 16h.
      v16h b = *(const v16h*)(bs + (t * 16 + lm) * 32 + 16 * hh);
      acc[2 * t + 0] = wmma16(a0, b, acc[2 * t + 0]);
      acc[2 * t + 1] = wmma16(a1, b, acc[2 * t + 1]);
    }
    __syncthreads(); // done with this buffer before TDM overwrites it
  }

  // C-layout: lane holds column n0+t*16+lm, rows m0+u*16+8h+r in VGPR r.
#pragma unroll
  for (int u = 0; u < 2; ++u)
#pragma unroll
    for (int t = 0; t < 4; ++t) {
      const int col = n0 + t * 16 + lm;
      const int hc = col >> 6, d = col & (cDH - 1);
#pragma unroll
      for (int r = 0; r < 8; ++r) {
        const int row = m0 + u * 16 + 8 * hh + r;
        const int b = row >> 11, s = row & (cS - 1);
        const float v = acc[2 * t + u][r];
        if (which == 0)
          Qo[(((size_t)(b * cH + hc)) * cS + s) * cDH + d] =
              (_Float16)(v * 0.125f);
        else if (which == 1)
          Ko[(((size_t)(b * cH + hc)) * cS + s) * cDH + d] = (_Float16)v;
        else
          Vt[(((size_t)(b * cH + hc)) * cDH + d) * cS + s] = (_Float16)v;
      }
    }
}

// ---------------------------------------------------------------------------
// Streaming causal attention: one wave per (b,h, 16-query tile).
// ---------------------------------------------------------------------------
__global__ __launch_bounds__(32) void attn_kernel(
    const _Float16* __restrict__ Qh, const _Float16* __restrict__ Kh,
    const _Float16* __restrict__ Vt, _Float16* __restrict__ Ao) {
  const int qt = blockIdx.x, bh = blockIdx.y;
  const int q0 = qt * 16;
  const _Float16* __restrict__ Qb = Qh + (size_t)bh * cS * cDH;
  const _Float16* __restrict__ Kb = Kh + (size_t)bh * cS * cDH;
  const _Float16* __restrict__ Vb = Vt + (size_t)bh * cDH * cS;
  const int l = threadIdx.x, lm = l & 15, hh = l >> 4;
  const int qglob = q0 + lm;

  // Q^T B-fragments (lane = query column), constant across the key loop.
  const v16h qb0 = *(const v16h*)(Qb + (size_t)(q0 + lm) * cDH + 16 * hh);
  const v16h qb1 = *(const v16h*)(Qb + (size_t)(q0 + lm) * cDH + 32 + 16 * hh);

  v8f o0 = {}, o1 = {}, o2 = {}, o3 = {};
  float mrun = -3.0e38f, lrun = 0.0f;

  for (int kb = 0; kb <= q0 + 15; kb += 32) {
    // logits^T tiles (keys kb..kb+15, kb+16..kb+31): A from K rows.
    const _Float16* kr0 = Kb + (size_t)(kb + lm) * cDH;
    const _Float16* kr1 = Kb + (size_t)(kb + 16 + lm) * cDH;
    v8f c0 = {}, c1 = {};
    c0 = wmma16(load_a_frag(kr0, 0, hh), qb0, c0);
    c0 = wmma16(load_a_frag(kr0, 32, hh), qb1, c0);
    c1 = wmma16(load_a_frag(kr1, 0, hh), qb0, c1);
    c1 = wmma16(load_a_frag(kr1, 32, hh), qb1, c1);

    // Causal mask (branch-free) + per-lane max over this lane's 16 keys.
    float mloc = -3.0e38f;
#pragma unroll
    for (int r = 0; r < 8; ++r) {
      const int key0 = kb + 8 * hh + r;
      const int key1 = kb + 16 + 8 * hh + r;
      float x = (key0 <= qglob) ? c0[r] : -3.0e38f;
      float y = (key1 <= qglob) ? c1[r] : -3.0e38f;
      c0[r] = x; c1[r] = y;
      mloc = fmaxf(mloc, fmaxf(x, y));
    }
    // Merge the two half-wave key subsets for this query, update running max.
    const float mpair = fmaxf(mloc, __shfl_xor(mloc, 16));
    const float mnew = fmaxf(mrun, mpair);
    const float alpha = __expf(mrun - mnew);

    // Probabilities pack directly into the next A-fragment:
    // A value i (0..7) = tile0 VGPR i, value 8+i = tile1 VGPR i.
    float sloc = 0.0f;
    v16h pa;
#pragma unroll
    for (int r = 0; r < 8; ++r) {
      float p = __expf(c0[r] - mnew);
      float q = __expf(c1[r] - mnew);
      sloc += p + q;
      pa[r] = (_Float16)p;
      pa[8 + r] = (_Float16)q;
    }
    lrun = lrun * alpha + (sloc + __shfl_xor(sloc, 16));
    mrun = mnew;

    // Rescale accumulators: broadcast alpha(q) into output C-layout rows.
#pragma unroll
    for (int r = 0; r < 8; ++r) {
      const float ar = __shfl(alpha, 8 * hh + r);
      o0[r] *= ar; o1[r] *= ar; o2[r] *= ar; o3[r] *= ar;
    }

    // V B-fragments from pre-transposed V: lane = dh column, keys kb+8h.. and kb+16+8h..
    const v16h vb0 = __builtin_shufflevector(*(const v8h*)(Vb + (size_t)(0 + lm) * cS + kb + 8 * hh), *(const v8h*)(Vb + (size_t)(0 + lm) * cS + kb + 16 + 8 * hh), 0, 1, 2, 3, 4, 5, 6, 7, 8, 9, 10, 11, 12, 13, 14, 15);
    const v16h vb1 = __builtin_shufflevector(*(const v8h*)(Vb + (size_t)(16 + lm) * cS + kb + 8 * hh), *(const v8h*)(Vb + (size_t)(16 + lm) * cS + kb + 16 + 8 * hh), 0, 1, 2, 3, 4, 5, 6, 7, 8, 9, 10, 11, 12, 13, 14, 15);
    const v16h vb2 = __builtin_shufflevector(*(const v8h*)(Vb + (size_t)(32 + lm) * cS + kb + 8 * hh), *(const v8h*)(Vb + (size_t)(32 + lm) * cS + kb + 16 + 8 * hh), 0, 1, 2, 3, 4, 5, 6, 7, 8, 9, 10, 11, 12, 13, 14, 15);
    const v16h vb3 = __builtin_shufflevector(*(const v8h*)(Vb + (size_t)(48 + lm) * cS + kb + 8 * hh), *(const v8h*)(Vb + (size_t)(48 + lm) * cS + kb + 16 + 8 * hh), 0, 1, 2, 3, 4, 5, 6, 7, 8, 9, 10, 11, 12, 13, 14, 15);
    o0 = wmma16(pa, vb0, o0);
    o1 = wmma16(pa, vb1, o1);
    o2 = wmma16(pa, vb2, o2);
    o3 = wmma16(pa, vb3, o3);
  }

  // Normalize and store f16 output tile into [B,S,H*DH].
  const int b = bh >> 4, hc = bh & 15;
#pragma unroll
  for (int r = 0; r < 8; ++r) {
    const float inv = 1.0f / __shfl(lrun, 8 * hh + r);
    const int row = q0 + 8 * hh + r;
    _Float16* dst = Ao + ((size_t)(b * cS + row)) * cD + hc * cDH;
    dst[0 + lm]  = (_Float16)(o0[r] * inv);
    dst[16 + lm] = (_Float16)(o1[r] * inv);
    dst[32 + lm] = (_Float16)(o2[r] * inv);
    dst[48 + lm] = (_Float16)(o3[r] * inv);
  }
}

// ---------------------------------------------------------------------------
// Output projection: O @ Wo^T -> f32, same TDM-staged structure as qkv_gemm.
// ---------------------------------------------------------------------------
__global__ __launch_bounds__(128) void out_proj(const _Float16* __restrict__ Ah,
                                                const _Float16* __restrict__ Woh,
                                                float* __restrict__ Out) {
  __shared__ alignas(64) _Float16 bsm[2 * 64 * 32];

  const int mblk = blockIdx.x * 128, n0 = blockIdx.y * 64;
  const int wave = threadIdx.x >> 5, l = threadIdx.x & 31;
  const int lm = l & 15, hh = l >> 4;
  const int m0 = mblk + wave * 32;
  const _Float16* __restrict__ arow0 = Ah + (size_t)(m0 + lm) * cD;
  const _Float16* __restrict__ arow1 = Ah + (size_t)(m0 + 16 + lm) * cD;
  const bool issuer = (threadIdx.x < 32);
  const _Float16* wtile = Woh + (size_t)n0 * cD;

  if (issuer) tdm_load_tile(bsm, wtile, 64u, (unsigned)cD);

  v8f acc[8] = {};
  constexpr int KSTEPS = cD / 32;
  for (int i = 0; i < KSTEPS; ++i) {
    const int k0 = i * 32;
    if (issuer) {
      if (i + 1 < KSTEPS) {
        tdm_load_tile(bsm + (size_t)((i + 1) & 1) * 2048, wtile + k0 + 32, 64u,
                      (unsigned)cD);
        __builtin_amdgcn_s_wait_tensorcnt(1);
      } else {
        __builtin_amdgcn_s_wait_tensorcnt(0);
      }
    }
    __syncthreads();

    const _Float16* bs = bsm + (size_t)(i & 1) * 2048;
    if (k0 + 32 < cD) __builtin_prefetch(arow0 + k0 + 32, 0, 1);
    v16h a0 = load_a_frag(arow0, k0, hh);
    v16h a1 = load_a_frag(arow1, k0, hh);
#pragma unroll
    for (int t = 0; t < 4; ++t) {
      v16h b = *(const v16h*)(bs + (t * 16 + lm) * 32 + 16 * hh);
      acc[2 * t + 0] = wmma16(a0, b, acc[2 * t + 0]);
      acc[2 * t + 1] = wmma16(a1, b, acc[2 * t + 1]);
    }
    __syncthreads();
  }

#pragma unroll
  for (int u = 0; u < 2; ++u)
#pragma unroll
    for (int t = 0; t < 4; ++t)
#pragma unroll
      for (int r = 0; r < 8; ++r)
        Out[(size_t)(m0 + u * 16 + 8 * hh + r) * cD + n0 + t * 16 + lm] =
            acc[2 * t + u][r];
}

// ---------------------------------------------------------------------------
extern "C" void kernel_launch(void* const* d_in, const int* in_sizes, int n_in,
                              void* d_out, int out_size, void* d_ws,
                              size_t ws_size, hipStream_t stream) {
  const float* X  = (const float*)d_in[0];
  const float* Wq = (const float*)d_in[1];
  const float* Wk = (const float*)d_in[2];
  const float* Wv = (const float*)d_in[3];
  const float* Wo = (const float*)d_in[4];

  char* ws = (char*)d_ws;
  // Workspace map (f16, total 48 MB):
  _Float16* Xh  = (_Float16*)(ws + (size_t)0);          //  8 MB [4096,1024]
  _Float16* Wqh = (_Float16*)(ws + ((size_t)8  << 20)); //  2 MB
  _Float16* Wkh = (_Float16*)(ws + ((size_t)10 << 20)); //  2 MB
  _Float16* Wvh = (_Float16*)(ws + ((size_t)12 << 20)); //  2 MB
  _Float16* Woh = (_Float16*)(ws + ((size_t)14 << 20)); //  2 MB
  _Float16* Qh  = (_Float16*)(ws + ((size_t)16 << 20)); //  8 MB [B,H,S,DH]
  _Float16* Kh  = (_Float16*)(ws + ((size_t)24 << 20)); //  8 MB [B,H,S,DH]
  _Float16* Vt  = (_Float16*)(ws + ((size_t)32 << 20)); //  8 MB [B,H,DH,S]
  _Float16* Ao  = (_Float16*)(ws + ((size_t)40 << 20)); //  8 MB [4096,1024]

  const int nX = cM * cD, nW = cD * cD;
  f32_to_f16<<<(nX + 255) / 256, 256, 0, stream>>>(X, Xh, nX);
  f32_to_f16<<<(nW + 255) / 256, 256, 0, stream>>>(Wq, Wqh, nW);
  f32_to_f16<<<(nW + 255) / 256, 256, 0, stream>>>(Wk, Wkh, nW);
  f32_to_f16<<<(nW + 255) / 256, 256, 0, stream>>>(Wv, Wvh, nW);
  f32_to_f16<<<(nW + 255) / 256, 256, 0, stream>>>(Wo, Woh, nW);

  qkv_gemm<<<dim3(cM / 128, cD / 64, 3), 128, 0, stream>>>(Xh, Wqh, Wkh, Wvh,
                                                           Qh, Kh, Vt);
  attn_kernel<<<dim3(cS / 16, cB * cH), 32, 0, stream>>>(Qh, Kh, Vt, Ao);
  out_proj<<<dim3(cM / 128, cD / 64), 128, 0, stream>>>(Ao, Woh, (float*)d_out);
}
